// MultiHeadAttention_10333691314666
// MI455X (gfx1250) — compile-verified, matrix operand order checked by reading, not run on hardware
//
#include <hip/hip_runtime.h>

// ---------------------------------------------------------------------------
// MI455X (gfx1250) multi-head causal attention, bf16 WMMA pipeline.
//   x[2,2048,1024] f32; Wq/Wk/Wv/Wo [1024,1024] f32; b_o[1024] f32
//   out[2,2048,1024] f32
// Round 2: GEMM waves now compute 64x64 tiles (16 WMMA per 8 fragment loads,
// 32 FLOP/byte from L2); attention waves process 32 query rows (two 16-row
// tiles sharing K/V fragments, 2:1 WMMA:load).
// ---------------------------------------------------------------------------

typedef __attribute__((ext_vector_type(16))) __bf16          v16bf;
typedef __attribute__((ext_vector_type(8)))  float           v8f;
typedef __attribute__((ext_vector_type(8)))  unsigned short  v8us;

#define DEVFN __device__ __forceinline__

union FragU { v16bf bf; v8us u[2]; };

// float -> bf16, round-to-nearest-even, bit-level
DEVFN __bf16 f2bf(float f) {
  union { float f; unsigned u; } x; x.f = f;
  unsigned r = (x.u + 0x7fffu + ((x.u >> 16) & 1u)) >> 16;
  union { unsigned short s; __bf16 b; } y; y.s = (unsigned short)r;
  return y.b;
}

// Load a 16x32 bf16 A-fragment (or 32x16 B-fragment from [N][K] storage).
// ISA layout: lanes 0-15 hold row (lane&15), K = {0..7, 16..23};
//             lanes 16-31 hold row (lane&15), K = {8..15, 24..31}.
// base must be 16B aligned, ld a multiple of 8 elements.
DEVFN v16bf load_frag(const __bf16* base, int ld) {
  int lane = threadIdx.x & 31;
  int r    = lane & 15;
  int kh   = (lane >> 4) << 3;            // 0 or 8
  const __bf16* p = base + r * ld + kh;
  FragU f;
  f.u[0] = *(const v8us*)(p);
  f.u[1] = *(const v8us*)(p + 16);
  return f.bf;
}

DEVFN v8f wmma_bf16(v16bf a, v16bf b, v8f c) {
  return __builtin_amdgcn_wmma_f32_16x16x32_bf16(
      /*neg_a=*/false, a, /*neg_b=*/false, b,
      /*c_mod=*/(short)0, c, /*reuse_a=*/false, /*reuse_b=*/false);
}

// ---------------------------------------------------------------------------
// Elementwise conversions
// ---------------------------------------------------------------------------
__global__ void k_cvt(const float* __restrict__ in, __bf16* __restrict__ out, int n) {
  int i = blockIdx.x * blockDim.x + threadIdx.x;
  if (i < n) out[i] = f2bf(in[i]);
}

// in: [1024][1024] row-major (K x N) f32 -> out: [N][K] bf16
__global__ void k_transpose(const float* __restrict__ in, __bf16* __restrict__ out) {
  int i = blockIdx.x * blockDim.x + threadIdx.x;     // 1024*1024 threads
  int k = i >> 10;
  int n = i & 1023;
  out[n * 1024 + k] = f2bf(in[i]);
}

// ---------------------------------------------------------------------------
// GEMM: C[4096x1024] = A[4096x1024](bf16) * Bt[1024(N)x1024(K)](bf16)
// Each wave computes a 64x64 tile via 4x4 accumulator fragments.
// mode 0: bf16 row-major store
// mode 2: bf16 store transposed-per-head: [b][h][d][t] (for V)
// mode 3: f32 store + bias (output projection)
// ---------------------------------------------------------------------------
__global__ __launch_bounds__(256) void k_gemm(const __bf16* __restrict__ A,
                                              const __bf16* __restrict__ Bt,
                                              __bf16* __restrict__ Ob,
                                              float* __restrict__ Of,
                                              const float* __restrict__ bias,
                                              int mode) {
  const int K = 1024;
  int wave = threadIdx.x >> 5;
  int lane = threadIdx.x & 31;
  int tile = blockIdx.x * 8 + wave;      // 128 blocks * 8 waves = 1024 tiles
  int tM = (tile & 63) << 6;             // 64 M-tiles of 64
  int tN = (tile >> 6) << 6;             // 16 N-tiles of 64

  v8f acc[4][4];
#pragma unroll
  for (int i = 0; i < 4; ++i)
#pragma unroll
    for (int j = 0; j < 4; ++j) acc[i][j] = (v8f){};

  for (int k0 = 0; k0 < K; k0 += 32) {
    v16bf a[4], b[4];
#pragma unroll
    for (int i = 0; i < 4; ++i)
      a[i] = load_frag(A + (size_t)(tM + 16 * i) * K + k0, K);
#pragma unroll
    for (int j = 0; j < 4; ++j)
      b[j] = load_frag(Bt + (size_t)(tN + 16 * j) * K + k0, K);
#pragma unroll
    for (int i = 0; i < 4; ++i)
#pragma unroll
      for (int j = 0; j < 4; ++j)
        acc[i][j] = wmma_bf16(a[i], b[j], acc[i][j]);
  }

  int colh   = lane & 15;
  int rowoff = (lane >> 4) << 3;         // 0 or 8

#pragma unroll
  for (int i = 0; i < 4; ++i) {
#pragma unroll
    for (int j = 0; j < 4; ++j) {
      int gr = tM + (i << 4);
      int gc = tN + (j << 4);
#pragma unroll
      for (int r = 0; r < 8; ++r) {
        int row = gr + rowoff + r;
        int col = gc + colh;
        float v = acc[i][j][r];
        if (mode == 0) {
          Ob[(size_t)row * 1024 + col] = f2bf(v);
        } else if (mode == 2) {
          // row = b*2048 + t ; col = h*64 + d ; Vt[((b*16+h)*64+d)*2048 + t]
          size_t vrow = (size_t)((row >> 11) << 10) + col;
          Ob[vrow * 2048 + (row & 2047)] = f2bf(v);
        } else {
          Of[(size_t)row * 1024 + col] = v + bias[col];
        }
      }
    }
  }
}

// ---------------------------------------------------------------------------
// Flash attention: one wave per 32-row query tile (two 16-row WMMA tiles
// sharing K/V fragments), causal, online softmax.
// Q,K: bf16 row-major [b*T+t][1024] (head h occupies cols h*64..h*64+63)
// Vt:  bf16 [ (b*16+h)*64 + d ][ T ]
// ctx: bf16 row-major [b*T+t][1024]
// ---------------------------------------------------------------------------
__global__ __launch_bounds__(128) void k_attn(const __bf16* __restrict__ Q,
                                              const __bf16* __restrict__ Km,
                                              const __bf16* __restrict__ Vt,
                                              __bf16* __restrict__ ctx) {
  // P staging: per wave two 16x32 bf16 tiles (2 KB); 4 waves -> 8 KB
  __shared__ __attribute__((aligned(16))) __bf16 lds_p[4 * 1024];
  const int T = 2048;
  int wave = threadIdx.x >> 5;
  int lane = threadIdx.x & 31;
  int id = blockIdx.x * 4 + wave;        // 512 blocks * 4 waves = 2048 tiles
  int qt = id & 63;                      // 32-row query tile within (b,h)
  int h  = (id >> 6) & 15;
  int b  = id >> 10;
  int qbase = qt << 5;

  const float SCALE = 0.125f;            // 1/sqrt(64)
  int colh   = lane & 15;
  int rowoff = (lane >> 4) << 3;

  const __bf16* qp = Q + ((size_t)(b * T + qbase)) * 1024 + (h << 6);
  v16bf aq[2][2];
#pragma unroll
  for (int t = 0; t < 2; ++t) {
    aq[t][0] = load_frag(qp + (size_t)(16 * t) * 1024, 1024);       // d 0..31
    aq[t][1] = load_frag(qp + (size_t)(16 * t) * 1024 + 32, 1024);  // d 32..63
  }

  v8f o[2][4];
#pragma unroll
  for (int t = 0; t < 2; ++t)
#pragma unroll
    for (int n = 0; n < 4; ++n) o[t][n] = (v8f){};

  float mrun[2][8], lrun[2][8];
#pragma unroll
  for (int t = 0; t < 2; ++t)
#pragma unroll
    for (int r = 0; r < 8; ++r) { mrun[t][r] = -1e30f; lrun[t][r] = 0.0f; }

  __bf16* lp = lds_p + wave * 1024;
  const __bf16* vbase = Vt + ((size_t)(((b << 4) + h) << 6)) * 2048;

  for (int kv0 = 0; kv0 < qbase + 32; kv0 += 32) {
    const __bf16* kp = Km + ((size_t)(b * T + kv0)) * 1024 + (h << 6);
    v16bf kb[4];
    kb[0] = load_frag(kp, 1024);                            // kv 0..15, d 0..31
    kb[1] = load_frag(kp + 32, 1024);                       // kv 0..15, d 32..63
    kb[2] = load_frag(kp + (size_t)16 * 1024, 1024);        // kv 16..31, d 0..31
    kb[3] = load_frag(kp + (size_t)16 * 1024 + 32, 1024);   // kv 16..31, d 32..63

#pragma unroll
    for (int t = 0; t < 2; ++t) {
      v8f s0 = (v8f){}, s1 = (v8f){};
      s0 = wmma_bf16(aq[t][0], kb[0], s0);
      s0 = wmma_bf16(aq[t][1], kb[1], s0);
      s1 = wmma_bf16(aq[t][0], kb[2], s1);
      s1 = wmma_bf16(aq[t][1], kb[3], s1);

#pragma unroll
      for (int r = 0; r < 8; ++r) {
        int qrow = qbase + 16 * t + rowoff + r;
        float x0 = s0[r] * SCALE; if (kv0 + colh > qrow)      x0 = -1e30f;
        float x1 = s1[r] * SCALE; if (kv0 + 16 + colh > qrow) x1 = -1e30f;
        float mx = fmaxf(x0, x1);
#pragma unroll
        for (int off = 1; off < 16; off <<= 1)
          mx = fmaxf(mx, __shfl_xor(mx, off, 32));
        float mnew = fmaxf(mrun[t][r], mx);
        float p0 = __expf(x0 - mnew);
        float p1 = __expf(x1 - mnew);
        float rs = p0 + p1;
#pragma unroll
        for (int off = 1; off < 16; off <<= 1)
          rs += __shfl_xor(rs, off, 32);
        float a = __expf(mrun[t][r] - mnew);
        lrun[t][r] = lrun[t][r] * a + rs;
        mrun[t][r] = mnew;
        // fold rescale of running context into this loop (saves registers)
        o[t][0][r] *= a; o[t][1][r] *= a; o[t][2][r] *= a; o[t][3][r] *= a;
        // C-fragment layout -> row-major P[16][32] in LDS (wave-private)
        __bf16* pr = lp + (t << 9) + ((rowoff + r) << 5);
        pr[colh]      = f2bf(p0);
        pr[colh + 16] = f2bf(p1);
      }
    }
    // cross-lane LDS dependency within the wave: drain DS counter
    asm volatile("s_wait_dscnt 0" ::: "memory");
    v16bf pa0 = load_frag(lp, 32);         // P tile 0 as 16x32 A-fragment
    v16bf pa1 = load_frag(lp + 512, 32);   // P tile 1
#pragma unroll
    for (int n = 0; n < 4; ++n) {
      v16bf vb = load_frag(vbase + (size_t)(16 * n) * 2048 + kv0, 2048);
      o[0][n] = wmma_bf16(pa0, vb, o[0][n]);
      o[1][n] = wmma_bf16(pa1, vb, o[1][n]);
    }
    // DS ops are in-order within a wave: next iteration's ds_stores cannot
    // pass the ds_loads above.
  }

#pragma unroll
  for (int t = 0; t < 2; ++t) {
    __bf16* cb = ctx + ((size_t)(b * T + qbase + 16 * t)) * 1024 + (h << 6);
#pragma unroll
    for (int r = 0; r < 8; ++r) {
      float linv = 1.0f / lrun[t][r];
      size_t rowp = (size_t)(rowoff + r) * 1024;
      cb[rowp + colh]      = f2bf(o[t][0][r] * linv);
      cb[rowp + 16 + colh] = f2bf(o[t][1][r] * linv);
      cb[rowp + 32 + colh] = f2bf(o[t][2][r] * linv);
      cb[rowp + 48 + colh] = f2bf(o[t][3][r] * linv);
    }
  }
}

// ---------------------------------------------------------------------------
// Host-side launch
// ---------------------------------------------------------------------------
extern "C" void kernel_launch(void* const* d_in, const int* in_sizes, int n_in,
                              void* d_out, int out_size, void* d_ws, size_t ws_size,
                              hipStream_t stream) {
  (void)in_sizes; (void)n_in; (void)out_size; (void)ws_size;
  const float* x  = (const float*)d_in[0];
  const float* Wq = (const float*)d_in[1];
  const float* Wk = (const float*)d_in[2];
  const float* Wv = (const float*)d_in[3];
  const float* Wo = (const float*)d_in[4];
  const float* bo = (const float*)d_in[5];
  float* out = (float*)d_out;

  const size_t XN = 4096ull * 1024;   // x / Q / K / V / ctx element count
  const size_t WN = 1024ull * 1024;   // weight element count

  __bf16* ws  = (__bf16*)d_ws;        // ~48 MiB of bf16 staging
  __bf16* xb  = ws;
  __bf16* WqT = xb + XN;
  __bf16* WkT = WqT + WN;
  __bf16* WvT = WkT + WN;
  __bf16* WoT = WvT + WN;
  __bf16* Qb  = WoT + WN;
  __bf16* Kb  = Qb + XN;
  __bf16* Vt  = Kb + XN;
  __bf16* ctx = Vt + XN;

  k_cvt<<<(int)(XN / 256), 256, 0, stream>>>(x, xb, (int)XN);
  k_transpose<<<(int)(WN / 256), 256, 0, stream>>>(Wq, WqT);
  k_transpose<<<(int)(WN / 256), 256, 0, stream>>>(Wk, WkT);
  k_transpose<<<(int)(WN / 256), 256, 0, stream>>>(Wv, WvT);
  k_transpose<<<(int)(WN / 256), 256, 0, stream>>>(Wo, WoT);

  k_gemm<<<128, 256, 0, stream>>>(xb, WqT, Qb, nullptr, nullptr, 0);
  k_gemm<<<128, 256, 0, stream>>>(xb, WkT, Kb, nullptr, nullptr, 0);
  k_gemm<<<128, 256, 0, stream>>>(xb, WvT, Vt, nullptr, nullptr, 2);

  k_attn<<<512, 128, 0, stream>>>(Qb, Kb, Vt, ctx);

  k_gemm<<<128, 256, 0, stream>>>(ctx, WoT, nullptr, out, bo, 3);
}
